// SimCLRLoss_40578851012794
// MI455X (gfx1250) — matrix operand order repaired and checked by reading, not compiled, not run on hardware
//
#include <hip/hip_runtime.h>
#include <stdint.h>

// ---------------- constants ----------------
#define N_TOTAL 8192
#define HALF_N  4096
#define D       256
#define INV_TEMP 2.0f          // 1 / 0.5
#define NEG_FILL -1.0e9f

#define WAVES    4                 // waves per block
#define BLOCKT   (WAVES * 32)      // 128 threads
#define ROWS_PB  (WAVES * 16)      // 64 rows per block
#define NJT      (N_TOTAL / 16)    // 512 column tiles
#define BPITCH   264               // 256 + 8 bf16 pad -> conflict-free ds_load_b128

// ---------------- types ----------------
typedef __bf16  bf16_t;
typedef bf16_t  v16bf __attribute__((ext_vector_type(16)));
typedef float   v8f   __attribute__((ext_vector_type(8)));
typedef unsigned int v4u __attribute__((ext_vector_type(4)));

union Frag16 { v16bf v; v4u q[2]; };   // 32B: one WMMA 16-bit A/B fragment per lane

// round-to-nearest-even fp32 -> bf16 bit pattern
__device__ __forceinline__ unsigned short f32_to_bf16_rne(float x) {
    uint32_t u = __float_as_uint(x);
    u += 0x7FFFu + ((u >> 16) & 1u);
    return (unsigned short)(u >> 16);
}
__device__ __forceinline__ float bf16_bits_to_f32(unsigned short b) {
    return __uint_as_float(((uint32_t)b) << 16);
}

// wave32 xor-shuffle via ds_bpermute (EXEC all-ones at call sites)
__device__ __forceinline__ float shxor(float v, int mask) {
    const int lane = (int)(threadIdx.x & 31u);
    const int src  = lane ^ mask;
    return __int_as_float(__builtin_amdgcn_ds_bpermute(src << 2, __float_as_int(v)));
}

// ---------------- kernel 1: fp32 -> bf16 hi/lo split ----------------
__global__ __launch_bounds__(256)
void simclr_prep(const float* __restrict__ z1, const float* __restrict__ z2,
                 unsigned short* __restrict__ Zhi, unsigned short* __restrict__ Zlo) {
    const int i = blockIdx.x * blockDim.x + threadIdx.x;
    const int total = N_TOTAL * D;
    if (i >= total) return;
    const int halfelems = HALF_N * D;
    float v = (i < halfelems) ? z1[i] : z2[i - halfelems];
    unsigned short hb = f32_to_bf16_rne(v);
    float lo = v - bf16_bits_to_f32(hb);
    Zhi[i] = hb;
    Zlo[i] = f32_to_bf16_rne(lo);
}

// ---------------- kernel 2: zero the (poisoned) output ----------------
__global__ void simclr_init_out(float* out) { out[0] = 0.0f; }

// ---------------- async staging: 16x256 tile (hi+lo) -> LDS double buffer ----------
// Per plane: 16 rows x 32 chunks of 16B = 512 chunks; block has WAVES*32 lanes,
// each lane moves 512/(WAVES*32) chunks per plane with global_load_async_to_lds_b128.
__device__ __forceinline__ void stage_tile(const unsigned short* __restrict__ Zhi,
                                           const unsigned short* __restrict__ Zlo,
                                           uint32_t ldsHi, uint32_t ldsLo,
                                           int j0, int wave, int lane) {
#pragma unroll
    for (int i = 0; i < 512 / (WAVES * 32); ++i) {
        const int c   = i * (WAVES * 32) + wave * 32 + lane;
        const int row = c >> 5;          // 0..15
        const int c8  = c & 31;          // 16B chunk within row
        const uint32_t goff = (uint32_t)(((j0 + row) * D + c8 * 8) * 2);      // bytes
        const uint32_t eo   = (uint32_t)((row * BPITCH + c8 * 8) * 2);        // bytes
        asm volatile("global_load_async_to_lds_b128 %0, %1, %2"
                     :: "v"(ldsHi + eo), "v"(goff), "s"(Zhi) : "memory");
        asm volatile("global_load_async_to_lds_b128 %0, %1, %2"
                     :: "v"(ldsLo + eo), "v"(goff), "s"(Zlo) : "memory");
    }
}

// ---------------- kernel 3: fused  z z^T/T + masked logsumexp + CE ----------------
// 128 blocks x 128 threads (4 waves). Wave w owns rows [blk*64 + w*16, +16) and
// walks ALL 512 column tiles; B tiles are staged into LDS once per block (async,
// double buffered) and consumed by all 4 waves via conflict-free ds_load_b128.
// Two WMMA accumulators (even/odd K-chunks) break the single RAW chain through C.
__global__ __launch_bounds__(BLOCKT)
void simclr_main(const unsigned short* __restrict__ Zhi,
                 const unsigned short* __restrict__ Zlo,
                 float* __restrict__ out) {
    __shared__ unsigned short BufHi[2][16 * BPITCH];   // 2 x 8448 B
    __shared__ unsigned short BufLo[2][16 * BPITCH];   // 2 x 8448 B
    __shared__ float lsum[2 * WAVES];

    const int t    = threadIdx.x;
    const int wave = t >> 5;
    const int lane = t & 31;
    const int h    = (lane >> 4) & 1;   // lane half: K sub-range selector
    const int lr   = lane & 15;         // row (A) / column (B) index in tile
    const int i0   = blockIdx.x * ROWS_PB + wave * 16;

    const uint32_t ldsHi0 = (uint32_t)(uintptr_t)&BufHi[0][0];
    const uint32_t ldsHi1 = (uint32_t)(uintptr_t)&BufHi[1][0];
    const uint32_t ldsLo0 = (uint32_t)(uintptr_t)&BufLo[0][0];
    const uint32_t ldsLo1 = (uint32_t)(uintptr_t)&BufLo[1][0];

    // ---- preload this wave's full A tile (hi+lo, K=256) into 128 VGPRs ----
    const unsigned short* arow_hi = Zhi + (size_t)(i0 + lr) * D;
    const unsigned short* arow_lo = Zlo + (size_t)(i0 + lr) * D;
    Frag16 Ahi[8], Alo[8];
#pragma unroll
    for (int k8 = 0; k8 < 8; ++k8) {
        const int aoff = k8 * 32 + 8 * h;    // A 16x32 layout: {8h..8h+7} U {16+8h..}
        Ahi[k8].q[0] = *(const v4u*)(arow_hi + aoff);
        Ahi[k8].q[1] = *(const v4u*)(arow_hi + aoff + 16);
        Alo[k8].q[0] = *(const v4u*)(arow_lo + aoff);
        Alo[k8].q[1] = *(const v4u*)(arow_lo + aoff + 16);
    }

    // online softmax state: C VGPR r holds row (r + 8h), col lr
    float m[8], s[8], tv[8];
#pragma unroll
    for (int r = 0; r < 8; ++r) { m[r] = -3.0e38f; s[r] = 0.0f; tv[r] = 0.0f; }

    // prime the pipeline: stage tile 0 into buffer 0
    stage_tile(Zhi, Zlo, ldsHi0, ldsLo0, 0, wave, lane);

    for (int jt = 0; jt < NJT; ++jt) {
        const int cur = jt & 1;
        const int j0  = jt * 16;

        asm volatile("s_wait_asynccnt 0x0" ::: "memory");  // my staging arrived
        __syncthreads();                                   // everyone's staging + prev compute done

        if (jt + 1 < NJT) {                                // prefetch next tile
            stage_tile(Zhi, Zlo,
                       (cur ? ldsHi0 : ldsHi1), (cur ? ldsLo0 : ldsLo1),
                       j0 + 16, wave, lane);
        }

        // two independent accumulator chains: even K-chunks -> c0, odd -> c1
        v8f c0 = {0.f, 0.f, 0.f, 0.f, 0.f, 0.f, 0.f, 0.f};
        v8f c1 = {0.f, 0.f, 0.f, 0.f, 0.f, 0.f, 0.f, 0.f};
#pragma unroll
        for (int k8 = 0; k8 < 8; ++k8) {
            // B 32x16 layout: lane half h covers K = {32k8+8h..+7} U {32k8+16+8h..+7}
            const int boff = k8 * 32 + 8 * h;
            Frag16 bhi, blo;
            const unsigned short* bh = &BufHi[cur][lr * BPITCH + boff];
            const unsigned short* bl = &BufLo[cur][lr * BPITCH + boff];
            bhi.q[0] = *(const v4u*)(bh);
            bhi.q[1] = *(const v4u*)(bh + 16);
            blo.q[0] = *(const v4u*)(bl);
            blo.q[1] = *(const v4u*)(bl + 16);
            if ((k8 & 1) == 0) {
                c0 = __builtin_amdgcn_wmma_f32_16x16x32_bf16(false, Ahi[k8].v, false, bhi.v,
                                                             (short)0, c0, false, false);
                c0 = __builtin_amdgcn_wmma_f32_16x16x32_bf16(false, Ahi[k8].v, false, blo.v,
                                                             (short)0, c0, false, false);
                c0 = __builtin_amdgcn_wmma_f32_16x16x32_bf16(false, Alo[k8].v, false, bhi.v,
                                                             (short)0, c0, false, false);
            } else {
                c1 = __builtin_amdgcn_wmma_f32_16x16x32_bf16(false, Ahi[k8].v, false, bhi.v,
                                                             (short)0, c1, false, false);
                c1 = __builtin_amdgcn_wmma_f32_16x16x32_bf16(false, Ahi[k8].v, false, blo.v,
                                                             (short)0, c1, false, false);
                c1 = __builtin_amdgcn_wmma_f32_16x16x32_bf16(false, Alo[k8].v, false, bhi.v,
                                                             (short)0, c1, false, false);
            }
        }

        // fused epilogue: temperature, diag mask, target capture, online logsumexp
        const int gj = j0 + lr;
#pragma unroll
        for (int r = 0; r < 8; ++r) {
            const int gi = i0 + r + 8 * h;
            float v = (c0[r] + c1[r]) * INV_TEMP;
            if (gj == gi) v = NEG_FILL;
            const int tgt = (gi < HALF_N) ? gi + HALF_N : gi - HALF_N;
            if (gj == tgt) tv[r] += v;       // exactly one (lane, tile) hit per row
            float nm = fmaxf(m[r], v);
            s[r] = s[r] * __expf(m[r] - nm) + __expf(v - nm);
            m[r] = nm;
        }
    }

    // reduce across the 16 lanes of each half (columns of a row)
#pragma unroll
    for (int r = 0; r < 8; ++r) {
#pragma unroll
        for (int o = 8; o >= 1; o >>= 1) {
            float mo = shxor(m[r], o);
            float so = shxor(s[r], o);
            float to = shxor(tv[r], o);
            float nm = fmaxf(m[r], mo);
            s[r] = s[r] * __expf(m[r] - nm) + so * __expf(mo - nm);
            m[r] = nm;
            tv[r] += to;
        }
    }

    // each wave owns complete rows: finish its 16-row loss locally
    if (lr == 0) {
        float acc = 0.0f;
#pragma unroll
        for (int r = 0; r < 8; ++r)
            acc += (m[r] + __logf(s[r])) - tv[r];   // logsumexp - sim[target]
        lsum[wave * 2 + h] = acc;
    }
    __syncthreads();

    if (t == 0) {
        float acc = 0.0f;
#pragma unroll
        for (int k = 0; k < 2 * WAVES; ++k) acc += lsum[k];
        atomicAdd(out, acc * (1.0f / (float)N_TOTAL));
    }
}

// ---------------- launch ----------------
extern "C" void kernel_launch(void* const* d_in, const int* in_sizes, int n_in,
                              void* d_out, int out_size, void* d_ws, size_t ws_size,
                              hipStream_t stream) {
    const float* z1 = (const float*)d_in[0];
    const float* z2 = (const float*)d_in[1];
    float* out = (float*)d_out;

    unsigned short* Zhi = (unsigned short*)d_ws;                      // 4 MB
    unsigned short* Zlo = Zhi + (size_t)N_TOTAL * D;                  // 4 MB

    const int total = N_TOTAL * D;
    simclr_prep<<<(total + 255) / 256, 256, 0, stream>>>(z1, z2, Zhi, Zlo);
    simclr_init_out<<<1, 1, 0, stream>>>(out);
    simclr_main<<<N_TOTAL / ROWS_PB, BLOCKT, 0, stream>>>(Zhi, Zlo, out);
}
